// SelfAttention_10883447128305
// MI455X (gfx1250) — compile-verified, matrix operand order checked by reading, not run on hardware
//
#include <hip/hip_runtime.h>
#include <hip/hip_bf16.h>
#include <stdint.h>

typedef __bf16 bf16_t;
typedef __attribute__((ext_vector_type(16))) __bf16 v16bf;
typedef __attribute__((ext_vector_type(8)))  __bf16 v8bf;
typedef __attribute__((ext_vector_type(8)))  float  v8f;

#define NH   16
#define HD   64
#define SEQ  2048
#define BATCH 2
#define DM   1024

#if defined(__has_builtin)
#  if __has_builtin(__builtin_amdgcn_tensor_load_to_lds)
#    define HAVE_TDM 1
#  endif
#endif
#ifndef HAVE_TDM
#  define HAVE_TDM 0
#endif

__device__ __forceinline__ bf16_t f2bf(float f) {
    union { float f; uint32_t u; } v; v.f = f;
    uint32_t u = v.u;
    uint32_t r = (u + 0x7FFFu + ((u >> 16) & 1u)) >> 16;
    union { uint16_t s; bf16_t b; } o; o.s = (uint16_t)r;
    return o.b;
}

__device__ __forceinline__ v8f wmma_bf16(v16bf a, v16bf b, v8f c) {
    // D = A(16x32 bf16) * B(32x16 bf16) + C(16x16 f32)
    return __builtin_amdgcn_wmma_f32_16x16x32_bf16(false, a, false, b,
                                                   (short)0, c, false, false);
}

union Frag { v16bf v; v8bf h[2]; };

#if HAVE_TDM
typedef __attribute__((ext_vector_type(4))) unsigned int v4u;
typedef __attribute__((ext_vector_type(8))) int v8i;
typedef __attribute__((ext_vector_type(4))) int v4i;

// TDM: DMA one 128-row x 32-col bf16 tile (row stride = strideElems) into LDS.
// D# per CDNA5 ISA; groups 2/3 zero => tile_dim2/3/4 unused.
// 6-arg builtin form (clang-23 / therock headers): g0, g1, g2, g3, g_extra, cpol.
__device__ __forceinline__ void tdm_load_tile(uint32_t ldsOff, const bf16_t* gsrc,
                                              int strideElems)
{
    const uint64_t ga = (uint64_t)(uintptr_t)gsrc;
    v4u g0;
    g0.x = 1u;                                                  // count=1, user mode
    g0.y = ldsOff;                                              // lds_addr (bytes)
    g0.z = (uint32_t)ga;                                        // global_addr[31:0]
    g0.w = (uint32_t)((ga >> 32) & 0x01FFFFFFu) | 0x80000000u;  // addr[56:32] | type=2
    v8i g1;
    g1[0] = 1 << 16;                                            // data_size = 2 bytes
    g1[1] = (int)(((uint32_t)strideElems & 0xFFFFu) << 16);     // tensor_dim0[15:0]
    g1[2] = (int)((((uint32_t)strideElems >> 16) & 0xFFFFu)     // tensor_dim0[31:16]
                  | (0xFFFFu << 16));                           // tensor_dim1[15:0]
    g1[3] = (int)(32u << 16);                                   // tensor_dim1 hi=0 | tile_dim0=32
    g1[4] = 128;                                                // tile_dim1=128, tile_dim2=0
    g1[5] = strideElems;                                        // tensor_dim0_stride[31:0]
    g1[6] = 0;                                                  // stride0 hi | stride1 lo
    g1[7] = 0;
    const v4i zero4 = {0, 0, 0, 0};
    const v8i zero8 = {0, 0, 0, 0, 0, 0, 0, 0};
    __builtin_amdgcn_tensor_load_to_lds(g0, g1, zero4, zero4, zero8, 0);
}
#endif

// ---------------------------------------------------------------------------
// elementwise f32 -> bf16
__global__ void cast_f32_bf16(const float* __restrict__ in, bf16_t* __restrict__ out, int n) {
    int i = blockIdx.x * blockDim.x + threadIdx.x;
    if (i < n) out[i] = f2bf(in[i]);
}

// in [K][N] f32 -> out [N][K] bf16 (weights pre-transposed so WMMA B-fragments
// read contiguous 16B chunks, matching the ISA B-matrix lane layout)
__global__ void transpose_cast(const float* __restrict__ in, bf16_t* __restrict__ out,
                               int K, int N) {
    int i = blockIdx.x * blockDim.x + threadIdx.x;
    if (i < K * N) {
        int n = i / K;
        int k = i - n * K;
        out[(size_t)n * K + k] = f2bf(in[(size_t)k * N + n]);
    }
}

// ---------------------------------------------------------------------------
// Tiled bf16 GEMM: C[M,N] = A[M,K] @ Bt[N,K]^T (+bias)
// Block tile 128x128x32, 8 waves (2 M x 4 N), 4x2 WMMA accumulators per wave.
// Tiles are fetched by the Tensor Data Mover (wave 0 issues 2 descriptors per
// k-step; TENSORcnt + barrier orders them) so all VALU/LDS issue goes to WMMA.
// mode 0: QKV epilogue  -> Q[bh][s][d], K[bh][s][d], V^T[bh][d][s]  (bf16)
// mode 1: output proj   -> Out[m][n] f32
__global__ __launch_bounds__(256) void gemm_bf16_128(
    const bf16_t* __restrict__ A, const bf16_t* __restrict__ Bt,
    const float* __restrict__ bias, int M, int N, int K, int mode,
    bf16_t* __restrict__ Qb, bf16_t* __restrict__ Kb, bf16_t* __restrict__ Vt,
    float* __restrict__ Out)
{
    __shared__ bf16_t As[2][128 * 32];
    __shared__ bf16_t Bs[2][128 * 32];

    const int tid  = threadIdx.x;
    const int lane = tid & 31;
    const int wave = tid >> 5;
    const int wm = wave >> 2;          // 0..1
    const int wn = wave & 3;           // 0..3
    const int m0 = blockIdx.y * 128;
    const int n0 = blockIdx.x * 128;
    const int r16 = lane & 15;
    const int kb  = (lane >> 4) << 3;  // 0 or 8

#if HAVE_TDM
    auto loadTiles = [&](int buf, int k0) {
        if (wave == 0) {
            tdm_load_tile((uint32_t)(uintptr_t)&As[buf][0],
                          A + (size_t)m0 * K + k0, K);
            tdm_load_tile((uint32_t)(uintptr_t)&Bs[buf][0],
                          Bt + (size_t)n0 * K + k0, K);
        }
    };
    auto waitTiles = [&]() {
        if (wave == 0) __builtin_amdgcn_s_wait_tensorcnt(0);
        __syncthreads();
    };
#else
    const uint32_t ldsA = (uint32_t)(uintptr_t)&As[0][0];
    const uint32_t ldsB = (uint32_t)(uintptr_t)&Bs[0][0];
    auto loadTiles = [&](int buf, int k0) {
        #pragma unroll
        for (int j = 0; j < 2; ++j) {
            int chunk = tid + j * 256;       // 512 chunks of 8 bf16 (16B)
            int row = chunk >> 2;            // 0..127
            int col = (chunk & 3) << 3;      // 0,8,16,24
            uint32_t la = ldsA + (uint32_t)buf * 8192u + (uint32_t)(row * 32 + col) * 2u;
            uint32_t lb = ldsB + (uint32_t)buf * 8192u + (uint32_t)(row * 32 + col) * 2u;
            uint64_t gaA = (uint64_t)(uintptr_t)(A  + (size_t)(m0 + row) * K + k0 + col);
            uint64_t gaB = (uint64_t)(uintptr_t)(Bt + (size_t)(n0 + row) * K + k0 + col);
            asm volatile("global_load_async_to_lds_b128 %0, %1, off"
                         :: "v"(la), "v"(gaA) : "memory");
            asm volatile("global_load_async_to_lds_b128 %0, %1, off"
                         :: "v"(lb), "v"(gaB) : "memory");
        }
    };
    auto waitTiles = [&]() {
        asm volatile("s_wait_asynccnt 0" ::: "memory");
        __syncthreads();
    };
#endif

    v8f acc[4][2] = {};
    loadTiles(0, 0);
    waitTiles();

    for (int k0 = 0; k0 < K; k0 += 32) {
        const int buf = (k0 >> 5) & 1;
        if (k0 + 32 < K) loadTiles(buf ^ 1, k0 + 32);

        Frag af[4], bfr[2];
        #pragma unroll
        for (int i = 0; i < 4; ++i) {
            const int row = wm * 64 + i * 16 + r16;
            af[i].h[0] = *(const v8bf*)&As[buf][row * 32 + kb];
            af[i].h[1] = *(const v8bf*)&As[buf][row * 32 + kb + 16];
        }
        #pragma unroll
        for (int j = 0; j < 2; ++j) {
            const int nrow = wn * 32 + j * 16 + r16;
            bfr[j].h[0] = *(const v8bf*)&Bs[buf][nrow * 32 + kb];
            bfr[j].h[1] = *(const v8bf*)&Bs[buf][nrow * 32 + kb + 16];
        }
        #pragma unroll
        for (int i = 0; i < 4; ++i)
            #pragma unroll
            for (int j = 0; j < 2; ++j)
                acc[i][j] = wmma_bf16(af[i].v, bfr[j].v, acc[i][j]);

        waitTiles();
    }

    // epilogue (C layout: VGPR r -> row = r + 8*(lane/16); col = lane%16)
    #pragma unroll
    for (int i = 0; i < 4; ++i) {
        #pragma unroll
        for (int j = 0; j < 2; ++j) {
            const int nglob = n0 + wn * 32 + j * 16 + r16;
            const float bv = bias[nglob];
            #pragma unroll
            for (int r = 0; r < 8; ++r) {
                const int mglob = m0 + wm * 64 + i * 16 + r + ((lane >> 4) << 3);
                const float v = acc[i][j][r] + bv;
                if (mode == 0) {
                    const int bb = mglob >> 11;     // batch
                    const int ss = mglob & 2047;    // seq
                    const int sec = nglob >> 10;    // 0=Q 1=K 2=V
                    const int nn = nglob & 1023;
                    const int hh = nn >> 6;
                    const int dd = nn & 63;
                    const size_t hb = (size_t)(bb * NH + hh);
                    if (sec == 0)      Qb[(hb * SEQ + ss) * HD + dd] = f2bf(v);
                    else if (sec == 1) Kb[(hb * SEQ + ss) * HD + dd] = f2bf(v);
                    else               Vt[(hb * HD + dd) * SEQ + ss] = f2bf(v); // V^T
                } else {
                    Out[(size_t)mglob * N + nglob] = v;
                }
            }
        }
    }
}

// ---------------------------------------------------------------------------
// Flash attention: one wave per 16 query rows, streaming 32-key tiles with
// online softmax. Q/K row-major [s][64], V transposed [d][s].
__global__ __launch_bounds__(256) void flash_attn(
    const bf16_t* __restrict__ Qb, const bf16_t* __restrict__ Kb,
    const bf16_t* __restrict__ Vt, bf16_t* __restrict__ Ab)
{
    __shared__ bf16_t Pt[8][16 * 32];   // per-wave P transpose bounce

    const int tid  = threadIdx.x;
    const int lane = tid & 31;
    const int wave = tid >> 5;
    const int bh = blockIdx.y;
    const int b = bh >> 4, h = bh & 15;
    const int qbase = blockIdx.x * 128 + wave * 16;
    const bf16_t* Qh = Qb + (size_t)bh * SEQ * HD;
    const bf16_t* Kh = Kb + (size_t)bh * SEQ * HD;
    const bf16_t* Vh = Vt + (size_t)bh * HD * SEQ;
    const int r16   = lane & 15;
    const int kb    = (lane >> 4) << 3;
    const int half8 = (lane >> 4) << 3;

    // Q fragments: A-matrix 16x32, two k-steps covering d=0..63
    Frag qf[2];
    #pragma unroll
    for (int s = 0; s < 2; ++s) {
        const bf16_t* qp = Qh + (size_t)(qbase + r16) * HD + s * 32 + kb;
        qf[s].h[0] = *(const v8bf*)qp;
        qf[s].h[1] = *(const v8bf*)(qp + 16);
    }

    v8f o[4] = {};
    float mst[8], lst[8];
    #pragma unroll
    for (int r = 0; r < 8; ++r) { mst[r] = -1e30f; lst[r] = 0.f; }

    const int ntiles = (qbase + 16 + 31) >> 5;   // causal key-tile limit
    for (int jt = 0; jt < ntiles; ++jt) {
        const int j0 = jt << 5;

        // K^T B-fragments: lane holds key column j0+t*16+r16, k-slices of d
        Frag kf[2][2];
        #pragma unroll
        for (int t = 0; t < 2; ++t) {
            const bf16_t* kp = Kh + (size_t)(j0 + t * 16 + r16) * HD;
            #pragma unroll
            for (int s = 0; s < 2; ++s) {
                kf[t][s].h[0] = *(const v8bf*)(kp + s * 32 + kb);
                kf[t][s].h[1] = *(const v8bf*)(kp + s * 32 + kb + 16);
            }
        }

        v8f s0 = {}, s1 = {};
        s0 = wmma_bf16(qf[0].v, kf[0][0].v, s0);
        s0 = wmma_bf16(qf[1].v, kf[0][1].v, s0);
        s1 = wmma_bf16(qf[0].v, kf[1][0].v, s1);
        s1 = wmma_bf16(qf[1].v, kf[1][1].v, s1);

        float cr[8];
        #pragma unroll
        for (int r = 0; r < 8; ++r) {
            const int q = qbase + r + half8;
            float e0 = s0[r] * 0.125f;            // 1/sqrt(64)
            float e1 = s1[r] * 0.125f;
            if (j0 + r16 > q)      e0 = -1e30f;   // causal mask
            if (j0 + 16 + r16 > q) e1 = -1e30f;
            float rm = fmaxf(e0, e1);
            #pragma unroll
            for (int off = 1; off < 16; off <<= 1)
                rm = fmaxf(rm, __shfl_xor(rm, off, 32));
            const float mnew = fmaxf(mst[r], rm);
            const float p0 = __expf(e0 - mnew);
            const float p1 = __expf(e1 - mnew);
            float rs = p0 + p1;
            #pragma unroll
            for (int off = 1; off < 16; off <<= 1)
                rs += __shfl_xor(rs, off, 32);
            const float c = __expf(mst[r] - mnew);
            lst[r] = lst[r] * c + rs;
            mst[r] = mnew;
            cr[r] = c;
            const int prow = r + half8;
            Pt[wave][prow * 32 + r16]      = f2bf(p0);
            Pt[wave][prow * 32 + 16 + r16] = f2bf(p1);
        }

        __builtin_amdgcn_wave_barrier();
        // reload P in A-fragment layout
        Frag pf;
        pf.h[0] = *(const v8bf*)&Pt[wave][r16 * 32 + kb];
        pf.h[1] = *(const v8bf*)&Pt[wave][r16 * 32 + kb + 16];

        #pragma unroll
        for (int t = 0; t < 4; ++t) {
            #pragma unroll
            for (int r = 0; r < 8; ++r) o[t][r] *= cr[r];   // rescale running O
            Frag vf;   // V^T rows are exactly the B-fragment rows
            const bf16_t* vp = Vh + (size_t)(t * 16 + r16) * SEQ + j0 + kb;
            vf.h[0] = *(const v8bf*)vp;
            vf.h[1] = *(const v8bf*)(vp + 16);
            o[t] = wmma_bf16(pf.v, vf.v, o[t]);
        }
        __builtin_amdgcn_wave_barrier();
    }

    float inv[8];
    #pragma unroll
    for (int r = 0; r < 8; ++r) inv[r] = 1.0f / lst[r];
    #pragma unroll
    for (int t = 0; t < 4; ++t)
        #pragma unroll
        for (int r = 0; r < 8; ++r) {
            const int s = qbase + r + half8;
            const int col = h * 64 + t * 16 + r16;
            Ab[(size_t)(b * SEQ + s) * DM + col] = f2bf(o[t][r] * inv[r]);
        }
}

// ---------------------------------------------------------------------------
extern "C" void kernel_launch(void* const* d_in, const int* in_sizes, int n_in,
                              void* d_out, int out_size, void* d_ws, size_t ws_size,
                              hipStream_t stream) {
    (void)in_sizes; (void)n_in; (void)out_size; (void)ws_size;
    const float* x    = (const float*)d_in[0];
    const float* Wqkv = (const float*)d_in[1];
    const float* bqkv = (const float*)d_in[2];
    const float* Wo   = (const float*)d_in[3];
    const float* bo   = (const float*)d_in[4];
    float* out = (float*)d_out;

    char* ws = (char*)d_ws;
    size_t off = 0;
    auto alloc = [&](size_t bytes) {
        size_t r = off; off += (bytes + 255) & ~(size_t)255; return r;
    };
    bf16_t* Xb  = (bf16_t*)(ws + alloc((size_t)BATCH * SEQ * DM * 2));       // 8 MB
    bf16_t* Wt  = (bf16_t*)(ws + alloc((size_t)3 * DM * DM * 2));            // 6 MB
    bf16_t* Wot = (bf16_t*)(ws + alloc((size_t)DM * DM * 2));                // 2 MB
    bf16_t* Qb  = (bf16_t*)(ws + alloc((size_t)BATCH * NH * SEQ * HD * 2));  // 8 MB
    bf16_t* Kb  = (bf16_t*)(ws + alloc((size_t)BATCH * NH * SEQ * HD * 2));  // 8 MB
    bf16_t* Vt  = (bf16_t*)(ws + alloc((size_t)BATCH * NH * HD * SEQ * 2));  // 8 MB
    bf16_t* Ab  = (bf16_t*)(ws + alloc((size_t)BATCH * SEQ * DM * 2));       // 8 MB

    const int nX = BATCH * SEQ * DM;            // 4194304
    cast_f32_bf16<<<(nX + 255) / 256, 256, 0, stream>>>(x, Xb, nX);
    transpose_cast<<<(3 * DM * DM + 255) / 256, 256, 0, stream>>>(Wqkv, Wt, DM, 3 * DM);
    transpose_cast<<<(DM * DM + 255) / 256, 256, 0, stream>>>(Wo, Wot, DM, DM);

    // QKV projection: M=4096, N=3072, K=1024
    gemm_bf16_128<<<dim3(3 * DM / 128, BATCH * SEQ / 128), 256, 0, stream>>>(
        Xb, Wt, bqkv, BATCH * SEQ, 3 * DM, DM, 0, Qb, Kb, Vt, nullptr);

    // Flash attention: grid = (q-chunks of 128, batch*heads)
    flash_attn<<<dim3(SEQ / 128, BATCH * NH), 256, 0, stream>>>(Qb, Kb, Vt, Ab);

    // Output projection: M=4096, N=1024, K=1024 -> f32 d_out
    gemm_bf16_128<<<dim3(DM / 128, BATCH * SEQ / 128), 256, 0, stream>>>(
        Ab, Wot, bo, BATCH * SEQ, DM, DM, 1, nullptr, nullptr, nullptr, out);
}
